// MHA_45887430590486
// MI455X (gfx1250) — matrix operand order repaired and checked by reading, not compiled, not run on hardware
//
#include <hip/hip_runtime.h>
#include <hip/hip_bf16.h>

// ---------------------------------------------------------------------------
// MHA forward for gfx1250 (MI455X): all matmuls via v_wmma_f32_16x16x32_bf16.
//   Stage 1: Q/K/V = x@W.T + b, fused RoPE; Q,K -> [B,H,N,d], V -> [B,H,d,N]
//   Stage 2: flash-attention; K/V^T tiles DMA'd to LDS via Tensor Data Mover
//            (double-buffered, s_wait_tensorcnt), online softmax, WMMA PV
//   Stage 3: out = attn@Wo.T + bo, fp32
// B=4, N=2048, HID=1024, H=16, d=64.
// ---------------------------------------------------------------------------

typedef __attribute__((ext_vector_type(16))) __bf16 v16bf;
typedef __attribute__((ext_vector_type(8)))  float  v8f;

#define BATCH   4
#define SEQ     2048
#define HID     1024
#define NHEAD   16
#define DHEAD   64

#if __has_builtin(__builtin_amdgcn_tensor_load_to_lds) && \
    __has_builtin(__builtin_amdgcn_s_wait_tensorcnt)
#define HAVE_TDM 1
#else
#define HAVE_TDM 0
#endif

__device__ __forceinline__ __bf16 to_bf16(float f) {
    unsigned u = __builtin_bit_cast(unsigned, f);
    u += 0x7FFFu + ((u >> 16) & 1u);               // round-to-nearest-even
    return __builtin_bit_cast(__bf16, (unsigned short)(u >> 16));
}

// A-fragment pair base (16-bit A 16x32), contiguous k order: K = 16*half + 2r
__device__ __forceinline__ int a_pair_base(int r, int half) {
    return 2 * r + 16 * half;
}

#if HAVE_TDM
// Build a D# for a 2-D bf16 tile load (ISA §8.3/8.4) and issue TENSOR_LOAD_TO_LDS.
//   tensor: [tensor_d1][tensor_d0] elements, row stride = stride_d0 elements
//   tile:   [tile_d1][tile_d0] starting at gsrc; packed into LDS at lds_off with
//           padding pad_amount (code, DWORDs) every pad_interval (code) DWORDs.
// This toolchain's builtin takes 6 operands (g0, g1, g2, g3, extra, cpol).
__device__ __forceinline__ void tdm_load_tile_bf16(
    const void* gsrc, unsigned lds_off,
    unsigned tensor_d0, unsigned tensor_d1, unsigned stride_d0,
    unsigned tile_d0, unsigned tile_d1,
    unsigned pad_interval, unsigned pad_amount) {
    typedef __attribute__((ext_vector_type(4))) unsigned tdm_v4u;
    typedef __attribute__((ext_vector_type(8))) int      tdm_v8i;
    typedef __attribute__((ext_vector_type(4))) int      tdm_v4i;
    const unsigned long long ga = (unsigned long long)(uintptr_t)gsrc;
    tdm_v4u g0;
    g0[0] = 1u;                                          // count=1 valid D#
    g0[1] = lds_off;                                     // lds_addr (bytes)
    g0[2] = (unsigned)ga;                                // global_addr[31:0]
    g0[3] = (unsigned)((ga >> 32) & 0x01FFFFFFull)       // global_addr[56:32]
          | (2u << 30);                                  // type = 2 ("image")
    tdm_v8i g1;
    g1[0] = (int)((1u << 16)                             // data_size = 2 bytes
                | (1u << 20)                             // pad_enable
                | (pad_interval << 22) | (pad_amount << 25));
    g1[1] = (int)((tensor_d0 & 0xFFFFu) << 16);          // tensor_dim0 @ bit48
    g1[2] = (int)(((tensor_d0 >> 16) & 0xFFFFu) | ((tensor_d1 & 0xFFFFu) << 16));
    g1[3] = (int)(((tensor_d1 >> 16) & 0xFFFFu) | ((tile_d0 & 0xFFFFu) << 16));
    g1[4] = (int)(tile_d1 & 0xFFFFu);                    // tile_dim1 (tile_dim2=0)
    g1[5] = (int)stride_d0;                              // tensor_dim0_stride lo
    g1[6] = 0;
    g1[7] = 0;
    tdm_v4i z4 = {0, 0, 0, 0};                           // D# groups 2/3 (2-D: unused)
    tdm_v8i z8 = {0, 0, 0, 0, 0, 0, 0, 0};
    __builtin_amdgcn_tensor_load_to_lds(g0, g1, z4, z4, z8, 0);
}
#endif

// ---------------------------------------------------------------------------
// Templated WMMA GEMM: Y[m,n] = sum_k A[m,k]*W[n,k] + bias[n]
// MODE 0: A = fp32 x; bias + RoPE (proj 0/1); bf16 out; proj2 stored transposed
// MODE 1: A = bf16 ws; bias; fp32 out [M, HID]
// ---------------------------------------------------------------------------
template <int MODE>
__global__ __launch_bounds__(128)
void gemm_rope_kernel(const float* __restrict__ Xf, const __bf16* __restrict__ Xb,
                      const float* __restrict__ W0, const float* __restrict__ W1,
                      const float* __restrict__ W2,
                      const float* __restrict__ b0, const float* __restrict__ b1,
                      const float* __restrict__ b2,
                      __bf16* __restrict__ Y0, __bf16* __restrict__ Y1,
                      __bf16* __restrict__ Y2,
                      float* __restrict__ Yf) {
    __shared__ __bf16 As[64][36];
    __shared__ __bf16 Bs[64][36];

    const int tid  = threadIdx.x;
    const int wave = tid >> 5;
    const int lane = tid & 31;
    const int half = lane >> 4;
    const int lrow = lane & 15;

    const int tile_n = blockIdx.x * 64;
    const int tile_m = blockIdx.y * 64;

    const float* W    = W0;
    const float* bias = b0;
    __bf16*      Yb   = Y0;
    bool do_rope = false, v_trans = false;
    if constexpr (MODE == 0) {
        const int p = blockIdx.z;
        W    = (p == 0) ? W0 : (p == 1) ? W1 : W2;
        bias = (p == 0) ? b0 : (p == 1) ? b1 : b2;
        Yb   = (p == 0) ? Y0 : (p == 1) ? Y1 : Y2;
        do_rope = (p < 2);
        v_trans = (p == 2);
    }

    v8f acc[4] = {};

    for (int kk = 0; kk < HID; kk += 32) {
#pragma unroll
        for (int i = 0; i < 4; ++i) {
            const int slot = tid + i * 128;        // 0..511 (4-elem chunks)
            const int row  = slot >> 3;
            const int ko   = (slot & 7) << 2;
            if constexpr (MODE == 0) {
                const float* src = Xf + (size_t)(tile_m + row) * HID + kk + ko;
#pragma unroll
                for (int j = 0; j < 4; ++j) As[row][ko + j] = to_bf16(src[j]);
            } else {
                const __bf16* src = Xb + (size_t)(tile_m + row) * HID + kk + ko;
#pragma unroll
                for (int j = 0; j < 4; ++j) As[row][ko + j] = src[j];
            }
            const float* wsrc = W + (size_t)(tile_n + row) * HID + kk + ko;
#pragma unroll
            for (int j = 0; j < 4; ++j) Bs[row][ko + j] = to_bf16(wsrc[j]);
        }
        if (kk + 32 < HID) {   // CDNA5 global_prefetch_b8 on next K-tile
            __builtin_prefetch(W + (size_t)(tile_n + (tid >> 1)) * HID + kk + 32, 0, 0);
        }
        __syncthreads();

        v16bf afrag;
        const __bf16* arow = &As[wave * 16 + lrow][0];
#pragma unroll
        for (int r = 0; r < 8; ++r) {
            const int k = a_pair_base(r, half);
            afrag[2 * r]     = arow[k];
            afrag[2 * r + 1] = arow[k + 1];
        }
#pragma unroll
        for (int c = 0; c < 4; ++c) {
            v16bf bfrag;
            const __bf16* brow = &Bs[c * 16 + lrow][16 * half];
#pragma unroll
            for (int e = 0; e < 16; ++e) bfrag[e] = brow[e];
            acc[c] = __builtin_amdgcn_wmma_f32_16x16x32_bf16(
                false, afrag, false, bfrag, (short)0, acc[c], false, false);
        }
        __syncthreads();
    }

#pragma unroll
    for (int c = 0; c < 4; ++c) {
        const int n  = tile_n + c * 16 + lrow;
        const float bb = bias[n];
#pragma unroll
        for (int r = 0; r < 8; ++r) {
            const int m = tile_m + wave * 16 + 8 * half + r;
            float v = acc[c][r] + bb;
            if constexpr (MODE == 0) {
                if (do_rope) {
                    const int pos = m & (SEQ - 1);
                    const int dd  = n & (DHEAD - 1);
                    const int pr  = dd >> 1;
                    const float other = __shfl_xor(v, 1, 32);
                    const float inv = __powf(10000.0f, -(float)(2 * pr) * (1.0f / DHEAD));
                    const float th  = (float)pos * inv;
                    const float cs = __cosf(th), sn = __sinf(th);
                    v = (dd & 1) ? (other * sn + v * cs) : (v * cs - other * sn);
                }
                const int b  = m >> 11, nn = m & (SEQ - 1);
                const int h  = n >> 6,  dd = n & (DHEAD - 1);
                const size_t idx = v_trans
                    ? (((size_t)b * NHEAD + h) * DHEAD + dd) * SEQ + nn   // V^T
                    : (((size_t)b * NHEAD + h) * SEQ + nn) * DHEAD + dd;  // Q,K
                Yb[idx] = to_bf16(v);
            } else {
                Yf[(size_t)m * HID + n] = v;
            }
        }
    }
}

// ---------------------------------------------------------------------------
// Flash attention. Q,K in [BH, N, d]; V^T in [BH, d, N]. 32-key tiles,
// TDM double-buffered into LDS (fallback: manual staging).
// ---------------------------------------------------------------------------
__global__ __launch_bounds__(128)
void flash_attn_kernel(const __bf16* __restrict__ Qm, const __bf16* __restrict__ Km,
                       const __bf16* __restrict__ Vtm, __bf16* __restrict__ Om) {
    __shared__ __bf16 Kb[2][32][68];    // [buf][key][dim]   (row stride 68)
    __shared__ __bf16 Vb[2][64][34];    // [buf][dim][key]   (row stride 34)
    __shared__ __bf16 Pl[4][16][36];    // per-wave P tile for C->A re-layout

    const int tid  = threadIdx.x;
    const int wave = tid >> 5;
    const int lane = tid & 31;
    const int half = lane >> 4;
    const int lrow = lane & 15;

    const int bh = blockIdx.y;
    const int q0 = blockIdx.x * 64 + wave * 16;

    const __bf16* Q  = Qm  + (size_t)bh * SEQ * DHEAD;
    const __bf16* Kp = Km  + (size_t)bh * SEQ * DHEAD;
    const __bf16* Vt = Vtm + (size_t)bh * SEQ * DHEAD;   // [64][2048]

    // Q A-fragments (16x64), loaded once from global
    v16bf aq[2];
    const __bf16* qrow = Q + (size_t)(q0 + lrow) * DHEAD;
#pragma unroll
    for (int kb = 0; kb < 2; ++kb) {
#pragma unroll
        for (int r = 0; r < 8; ++r) {
            const int k = kb * 32 + a_pair_base(r, half);
            aq[kb][2 * r]     = qrow[k];
            aq[kb][2 * r + 1] = qrow[k + 1];
        }
    }

    v8f oacc[4] = {};
    float mstat[8], lstat[8];
#pragma unroll
    for (int r = 0; r < 8; ++r) { mstat[r] = -1e30f; lstat[r] = 0.0f; }

    const int NTILE = SEQ / 32;                          // 64 key tiles

#if HAVE_TDM
    const unsigned koff[2] = { (unsigned)(uintptr_t)&Kb[0][0][0],
                               (unsigned)(uintptr_t)&Kb[1][0][0] };
    const unsigned voff[2] = { (unsigned)(uintptr_t)&Vb[0][0][0],
                               (unsigned)(uintptr_t)&Vb[1][0][0] };
    if (wave == 0) {
        // K tile 0: tensor [2048 keys][64 d], contiguous dim0=64, pad 32DW->+2DW (stride 68)
        tdm_load_tile_bf16(Kp, koff[0], DHEAD, SEQ, DHEAD, DHEAD, 32, 4u, 1u);
        // V^T tile 0: tensor [64 d][2048 keys], contiguous dim0=2048, pad 16DW->+1DW (stride 34)
        tdm_load_tile_bf16(Vt, voff[0], SEQ, DHEAD, SEQ, 32, DHEAD, 3u, 0u);
    }
#endif

    for (int it = 0; it < NTILE; ++it) {
        const int j0  = it * 32;
        const int buf = it & 1;

        __syncthreads();   // everyone done reading the buffer we are about to fill
#if HAVE_TDM
        if (wave == 0 && it + 1 < NTILE) {
            const int jn = (it + 1) * 32;
            tdm_load_tile_bf16(Kp + (size_t)jn * DHEAD, koff[buf ^ 1],
                               DHEAD, SEQ, DHEAD, DHEAD, 32, 4u, 1u);
            tdm_load_tile_bf16(Vt + jn, voff[buf ^ 1],
                               SEQ, DHEAD, SEQ, 32, DHEAD, 3u, 0u);
        }
        if (it + 1 < NTILE) __builtin_amdgcn_s_wait_tensorcnt(2);
        else                __builtin_amdgcn_s_wait_tensorcnt(0);
#else
        // fallback: cooperative synchronous staging into buf
#pragma unroll
        for (int i = 0; i < 4; ++i) {
            const int slot = tid + i * 128;
            const int krow = slot >> 4;            // 16 chunks per 64-wide K row
            const int kko  = (slot & 15) << 2;
            const __bf16* ks = Kp + (size_t)(j0 + krow) * DHEAD + kko;
#pragma unroll
            for (int j = 0; j < 4; ++j) Kb[buf][krow][kko + j] = ks[j];
            const int vrow = slot >> 3;            // 8 chunks per 32-wide V^T row
            const int vko  = (slot & 7) << 2;
            const __bf16* vs = Vt + (size_t)vrow * SEQ + j0 + vko;
#pragma unroll
            for (int j = 0; j < 4; ++j) Vb[buf][vrow][vko + j] = vs[j];
        }
#endif
        __syncthreads();   // tile `it` visible to all waves

        // ---- scores S = Q K^T  (16 x 32) ----
        v8f sc[2];
#pragma unroll
        for (int t = 0; t < 2; ++t) {
            v8f z = {};
#pragma unroll
            for (int kb = 0; kb < 2; ++kb) {
                v16bf bk;
                const __bf16* kr = &Kb[buf][t * 16 + lrow][kb * 32 + 16 * half];
#pragma unroll
                for (int e = 0; e < 16; ++e) bk[e] = kr[e];
                z = __builtin_amdgcn_wmma_f32_16x16x32_bf16(
                    false, aq[kb], false, bk, (short)0, z, false, false);
            }
            sc[t] = z;
        }

        // ---- online softmax (row stats in the 16-lane half) ----
#pragma unroll
        for (int r = 0; r < 8; ++r) {
            const float s0 = sc[0][r] * 0.125f;    // 1/sqrt(64)
            const float s1 = sc[1][r] * 0.125f;
            float mx = fmaxf(s0, s1);
#pragma unroll
            for (int d = 1; d < 16; d <<= 1) mx = fmaxf(mx, __shfl_xor(mx, d, 16));
            const float nm   = fmaxf(mstat[r], mx);
            const float corr = __expf(mstat[r] - nm);
            const float p0 = __expf(s0 - nm), p1 = __expf(s1 - nm);
            float rs = p0 + p1;
#pragma unroll
            for (int d = 1; d < 16; d <<= 1) rs += __shfl_xor(rs, d, 16);
            lstat[r] = lstat[r] * corr + rs;
            mstat[r] = nm;
#pragma unroll
            for (int c = 0; c < 4; ++c) oacc[c][r] = oacc[c][r] * corr;
            const int prow = r + 8 * half;
            Pl[wave][prow][lrow]      = to_bf16(p0);
            Pl[wave][prow][16 + lrow] = to_bf16(p1);
        }

        // ---- re-layout P (C layout -> A fragment) via per-wave LDS ----
        v16bf ap;
        const __bf16* pr_ = &Pl[wave][lrow][0];
#pragma unroll
        for (int r = 0; r < 8; ++r) {
            const int k = a_pair_base(r, half);
            ap[2 * r]     = pr_[k];
            ap[2 * r + 1] = pr_[k + 1];
        }

        // ---- O += P V   (B fragment contiguous thanks to V^T) ----
#pragma unroll
        for (int c = 0; c < 4; ++c) {
            v16bf bv;
            const __bf16* vr = &Vb[buf][c * 16 + lrow][16 * half];
#pragma unroll
            for (int e = 0; e < 16; ++e) bv[e] = vr[e];
            oacc[c] = __builtin_amdgcn_wmma_f32_16x16x32_bf16(
                false, ap, false, bv, (short)0, oacc[c], false, false);
        }
    }

    // ---- normalize + store bf16 attn output in [B, N, H*d] ----
    const int b = bh >> 4, h = bh & 15;
#pragma unroll
    for (int r = 0; r < 8; ++r) {
        const int nn = q0 + 8 * half + r;
        const float invl = 1.0f / lstat[r];
#pragma unroll
        for (int c = 0; c < 4; ++c) {
            const size_t idx = ((size_t)b * SEQ + nn) * HID + h * DHEAD + c * 16 + lrow;
            Om[idx] = to_bf16(oacc[c][r] * invl);
        }
    }
}

// ---------------------------------------------------------------------------
extern "C" void kernel_launch(void* const* d_in, const int* in_sizes, int n_in,
                              void* d_out, int out_size, void* d_ws, size_t ws_size,
                              hipStream_t stream) {
    (void)in_sizes; (void)n_in; (void)out_size; (void)ws_size;
    const float* x  = (const float*)d_in[0];
    const float* Wq = (const float*)d_in[1];
    const float* bq = (const float*)d_in[2];
    const float* Wk = (const float*)d_in[3];
    const float* bk = (const float*)d_in[4];
    const float* Wv = (const float*)d_in[5];
    const float* bv = (const float*)d_in[6];
    const float* Wo = (const float*)d_in[7];
    const float* bo = (const float*)d_in[8];
    float* out = (float*)d_out;

    const size_t PER = (size_t)BATCH * NHEAD * SEQ * DHEAD;  // 8,388,608 bf16
    __bf16* q_ws = (__bf16*)d_ws;
    __bf16* k_ws = q_ws + PER;
    __bf16* v_ws = k_ws + PER;   // transposed [BH, d, N]
    __bf16* o_ws = v_ws + PER;

    dim3 blk(128);
    gemm_rope_kernel<0><<<dim3(16, 128, 3), blk, 0, stream>>>(
        x, nullptr, Wq, Wk, Wv, bq, bk, bv, q_ws, k_ws, v_ws, nullptr);
    flash_attn_kernel<<<dim3(32, 64), blk, 0, stream>>>(q_ws, k_ws, v_ws, o_ws);
    gemm_rope_kernel<1><<<dim3(16, 128, 1), blk, 0, stream>>>(
        nullptr, o_ws, Wo, nullptr, nullptr, bo, nullptr, nullptr,
        nullptr, nullptr, nullptr, out);
}
